// FP8Linear_54193897341245
// MI455X (gfx1250) — 8-bit matrix operand order repaired and checked by reading, compiled, not run on hardware
//
#include <hip/hip_runtime.h>
#include <hip/hip_bf16.h>


typedef __attribute__((ext_vector_type(16))) int   v16i;
typedef __attribute__((ext_vector_type(8)))  float v8f;

#define FP8_INV_MAX (1.0f / 448.0f)
#define QEPS 1e-12f

__device__ inline unsigned int f32_to_e4m3(float x) {
  unsigned int sign = (__float_as_uint(x) >> 31) << 7;
  float a = fabsf(x);
  if (!(a > 0.0f)) return sign;
  a = fminf(a, 448.0f);
  unsigned int bits = __float_as_uint(a);
  int e = (int)(bits >> 23) - 127;
  if (e < -6) {
    int m = (int)(a * 512.0f + 0.5f);
    if (m >= 8) return sign | 0x08;
    return sign | (unsigned int)m;
  }
  unsigned int mant = bits & 0x7FFFFFu;
  unsigned int lsb  = (mant >> 20) & 1u;
  mant += 0x7FFFFu + lsb;
  if (mant >> 23) { mant = 0; e += 1; }
  if (e > 8) return sign | 0x7E;
  unsigned int m3 = (mant >> 20) & 7u;
  if (e == 8 && m3 > 6u) m3 = 6u;
  return sign | ((unsigned int)(e + 7) << 3) | m3;
}

__device__ inline unsigned int pack4_e4m3(float x0, float x1, float x2, float x3) {
#if defined(__gfx1250__) && __has_builtin(__builtin_amdgcn_cvt_pk_fp8_f32)
  int p = 0;
  p = __builtin_amdgcn_cvt_pk_fp8_f32(x0, x1, p, false);
  p = __builtin_amdgcn_cvt_pk_fp8_f32(x2, x3, p, true);
  return (unsigned int)p;
#else
  return f32_to_e4m3(x0) | (f32_to_e4m3(x1) << 8) |
         (f32_to_e4m3(x2) << 16) | (f32_to_e4m3(x3) << 24);
#endif
}

__global__ __launch_bounds__(256) void amax_kernel(const float4* __restrict__ src,
                                                   int n4, unsigned int* __restrict__ out) {
  float m = 0.0f;
  for (int i = blockIdx.x * blockDim.x + threadIdx.x; i < n4;
       i += gridDim.x * blockDim.x) {
    float4 v = src[i];
    m = fmaxf(m, fabsf(v.x));
    m = fmaxf(m, fabsf(v.y));
    m = fmaxf(m, fabsf(v.z));
    m = fmaxf(m, fabsf(v.w));
  }
  __shared__ float red[256];
  red[threadIdx.x] = m;
  __syncthreads();
  for (int s = 128; s > 0; s >>= 1) {
    if ((int)threadIdx.x < s) red[threadIdx.x] = fmaxf(red[threadIdx.x], red[threadIdx.x + s]);
    __syncthreads();
  }
  if (threadIdx.x == 0) atomicMax(out, __float_as_uint(red[0]));
}

__global__ __launch_bounds__(256) void quant_kernel(const float4* __restrict__ src,
                                                    int n4,
                                                    const unsigned int* __restrict__ amax_bits,
                                                    unsigned int* __restrict__ dst) {
  float amax = __uint_as_float(*amax_bits);
  float inv  = 1.0f / fmaxf(amax * FP8_INV_MAX, QEPS);
  int i = blockIdx.x * blockDim.x + threadIdx.x;
  if (i < n4) {
    float4 v = src[i];
    dst[i] = pack4_e4m3(v.x * inv, v.y * inv, v.z * inv, v.w * inv);
  }
}

__device__ inline v16i load_a_tile(const unsigned char* __restrict__ ap) {
  v16i A;
#pragma unroll
  for (int t = 0; t < 2; ++t)
#pragma unroll
    for (int c = 0; c < 4; ++c) {
      unsigned long long d = *(const unsigned long long*)(ap + t * 64 + (c >> 1) * 32 + (c & 1) * 8);
      int i = (t * 4 + c) * 2;
      A[i]     = (int)(unsigned int)d;
      A[i + 1] = (int)(unsigned int)(d >> 32);
    }
  return A;
}

__device__ inline v16i load_b_tile(const unsigned char* __restrict__ bp) {
  v16i B;
#pragma unroll
  for (int g = 0; g < 4; ++g) {
    int4 q = *(const int4*)(bp + g * 32);
    B[g * 4 + 0] = q.x;
    B[g * 4 + 1] = q.y;
    B[g * 4 + 2] = q.z;
    B[g * 4 + 3] = q.w;
  }
  return B;
}

#define WMMA_FP8(Ar, Br, Cr) \
  __builtin_amdgcn_wmma_f32_16x16x128_fp8_fp8((Ar), (Br), (short)0, (Cr), false, false)

__global__ __launch_bounds__(256) void fp8_gemm_wmma(
    const unsigned char* __restrict__ Afp8,
    const unsigned char* __restrict__ Bfp8,
    const float* __restrict__ bias,
    const unsigned int* __restrict__ amax_bits,
    float* __restrict__ out,
    int N, int K) {
  const int lane = threadIdx.x & 31;
  const int wave = threadIdx.x >> 5;
  const int half = lane >> 4;
  const int l16  = lane & 15;

  const int mTile = (blockIdx.y * 8 + wave) * 32;
  const int nBase = blockIdx.x * 64;

  const float sx = fmaxf(__uint_as_float(amax_bits[0]) * FP8_INV_MAX, QEPS);
  const float sw = fmaxf(__uint_as_float(amax_bits[1]) * FP8_INV_MAX, QEPS);
  const float scale = sx * sw;

  const unsigned char* Ar0 = Afp8 + (size_t)(mTile + l16) * (size_t)K + (size_t)(half * 16);
  const unsigned char* Ar1 = Ar0 + (size_t)16 * (size_t)K;
  const unsigned char* Bb  = Bfp8 + (size_t)(nBase + l16) * (size_t)K + (size_t)(half * 16);
  const size_t bStride16 = (size_t)16 * (size_t)K;

  v8f acc[2][4] = {};

  v16i a0   = load_a_tile(Ar0);
  v16i a1   = load_a_tile(Ar1);
  v16i bcur = load_b_tile(Bb);
  v16i bnxt = load_b_tile(Bb + bStride16);

#pragma unroll 2
  for (int k0 = 0; k0 < K; k0 += 128) {
    int kn = k0 + 128;
    if (kn > K - 128) kn = K - 128;
    v16i na0 = load_a_tile(Ar0 + kn);
    v16i na1 = load_a_tile(Ar1 + kn);

#pragma unroll
    for (int s = 0; s < 4; ++s) {
      const int s2 = (s + 2) & 3;
      const int k2 = (s >= 2) ? kn : k0;
      v16i bpre = load_b_tile(Bb + (size_t)s2 * bStride16 + k2);

      acc[0][s] = WMMA_FP8(a0, bcur, acc[0][s]);
      acc[1][s] = WMMA_FP8(a1, bcur, acc[1][s]);
      bcur = bnxt;
      bnxt = bpre;
    }
    a0 = na0;
    a1 = na1;
  }

#pragma unroll
  for (int mt = 0; mt < 2; ++mt) {
#pragma unroll
    for (int s = 0; s < 4; ++s) {
      int n = nBase + s * 16 + l16;
      float bv = bias[n];
      float* orow = out + (size_t)(mTile + mt * 16 + half * 8) * (size_t)N + n;
#pragma unroll
      for (int v = 0; v < 8; ++v) {
        orow[(size_t)v * N] = acc[mt][s][v] * scale + bv;
      }
    }
  }
}

extern "C" void kernel_launch(void* const* d_in, const int* in_sizes, int n_in,
                              void* d_out, int out_size, void* d_ws, size_t ws_size,
                              hipStream_t stream) {
  const float* x    = (const float*)d_in[0];
  const float* w    = (const float*)d_in[1];
  const float* bias = (const float*)d_in[2];
  float* out = (float*)d_out;

  const int N = in_sizes[2];
  const int K = in_sizes[1] / N;
  const int M = in_sizes[0] / K;

  unsigned int*  amax = (unsigned int*)d_ws;
  unsigned char* xq   = (unsigned char*)d_ws + 256;
  unsigned char* wq   = xq + (size_t)M * (size_t)K;

  hipMemsetAsync(d_ws, 0, 8, stream);

  const int n4x = (M / 4) * K;
  const int n4w = (N / 4) * K;

  amax_kernel<<<2048, 256, 0, stream>>>((const float4*)x, n4x, amax + 0);
  amax_kernel<<<512, 256, 0, stream>>>((const float4*)w, n4w, amax + 1);

  quant_kernel<<<(n4x + 255) / 256, 256, 0, stream>>>((const float4*)x, n4x, amax + 0,
                                                      (unsigned int*)xq);
  quant_kernel<<<(n4w + 255) / 256, 256, 0, stream>>>((const float4*)w, n4w, amax + 1,
                                                      (unsigned int*)wq);

  dim3 grid(N / 64, M / 256);
  fp8_gemm_wmma<<<grid, 256, 0, stream>>>(xq, wq, bias, amax, out, N, K);
}
